// GRNN_38706245271761
// MI455X (gfx1250) — matrix operand order repaired and checked by reading, not compiled, not run on hardware
//
#include <hip/hip_runtime.h>
#include <hip/hip_bf16.h>
#include <stddef.h>

// ---------------------------------------------------------------------------
// GRNN forward for MI455X (gfx1250, wave32, WMMA).
//
//   out[i] = (sum_k basis_ik * s_k) / (sum_k basis_ik)
//   basis_ik = exp(-(x2_i + p2_k - 2*(x@W)_ik) / (2*sigma^2)),  sigma = 32
//   s_k  = sum_j W[k][j]   (row sums)   -> replaces the basis@W matmul
//   p2_k = sum_m W[m][k]^2 (col norms)
//
// Dominant cost: x@W = 137.4 GFLOP, x traffic = 256 MB (~11us @ 23.3 TB/s).
// f16 WMMA (16x16x32, f32 accum) -> ~27us compute: balanced/compute bound.
// basis is never materialized; exp + both reductions fused into the epilogue.
// ---------------------------------------------------------------------------

typedef __attribute__((ext_vector_type(16))) _Float16 v16h;
typedef __attribute__((ext_vector_type(8)))  _Float16 v8h;
typedef __attribute__((ext_vector_type(8)))  float    v8f;
typedef __attribute__((ext_vector_type(4)))  float    v4f;

#define D_DIM  1024
#define M_TILE 64
#define PITCH  1032   // halves per LDS row of x-tile: 1024 + 8 pad (16B) to
                      // spread the per-row stride across LDS banks

// -------------------------------------------------------------------------
// Prep 1: one block per row k of W.
//   s[k] = sum_j W[k][j];  Wt[n][k] = (f16)W[k][n]  (B-fragment friendly:
//   each WMMA lane then loads 16 contiguous K-halves = one 32B load).
// -------------------------------------------------------------------------
__global__ __launch_bounds__(256)
void grnn_prep_w(const float* __restrict__ W, _Float16* __restrict__ Wt,
                 float* __restrict__ s)
{
    __shared__ float part[8];
    const int k = blockIdx.x;
    const int t = threadIdx.x;

    v4f f = ((const v4f*)(W + (size_t)k * D_DIM))[t];   // 256*4 = 1024
    float acc = f[0] + f[1] + f[2] + f[3];
#pragma unroll
    for (int e = 0; e < 4; ++e)
        Wt[(size_t)(4 * t + e) * D_DIM + k] = (_Float16)f[e];

#pragma unroll
    for (int m = 1; m < 32; m <<= 1) acc += __shfl_xor(acc, m, 32);
    if ((t & 31) == 0) part[t >> 5] = acc;
    __syncthreads();
    if (t == 0) {
        float r = 0.f;
#pragma unroll
        for (int w = 0; w < 8; ++w) r += part[w];
        s[k] = r;
    }
}

// -------------------------------------------------------------------------
// Prep 2: p2[n] = sum_k Wt[n][k]^2 from the f16 copy (same rounding as the
// GEMM operand -> error partially cancels in d2 = x2 + p2 - 2*dot).
// -------------------------------------------------------------------------
__global__ __launch_bounds__(128)
void grnn_prep_p2(const _Float16* __restrict__ Wt, float* __restrict__ p2)
{
    __shared__ float part[4];
    const int n = blockIdx.x;
    const int t = threadIdx.x;

    v8h h = *(const v8h*)(Wt + (size_t)n * D_DIM + t * 8);  // 128*8 = 1024
    float acc = 0.f;
#pragma unroll
    for (int e = 0; e < 8; ++e) { float f = (float)h[e]; acc += f * f; }
#pragma unroll
    for (int m = 1; m < 32; m <<= 1) acc += __shfl_xor(acc, m, 32);
    if ((t & 31) == 0) part[t >> 5] = acc;
    __syncthreads();
    if (t == 0) p2[n] = part[0] + part[1] + part[2] + part[3];
}

// -------------------------------------------------------------------------
// Main fused kernel: one block = 64 rows of x, 8 wave32s tiled 4(M) x 2(N).
// -------------------------------------------------------------------------
__global__ __launch_bounds__(256)
void grnn_main(const float* __restrict__ x, const _Float16* __restrict__ Wt,
               const float* __restrict__ p2g, const float* __restrict__ sg,
               float* __restrict__ out)
{
    __shared__ _Float16 ldsX[M_TILE * PITCH];   // 129 KB f16 x-tile
    __shared__ float ldsX2[M_TILE];
    __shared__ float ldsP2[D_DIM];
    __shared__ float ldsS[D_DIM];
    __shared__ float ldsPart[256];
    __shared__ float ldsNum[M_TILE][2];
    __shared__ float ldsDen[M_TILE][2];

    const int t       = threadIdx.x;
    const int block_m = blockIdx.x * M_TILE;

    // p2 / s into LDS (1024 floats each, 4 per thread)
    ((v4f*)ldsP2)[t] = ((const v4f*)p2g)[t];
    ((v4f*)ldsS)[t]  = ((const v4f*)sg)[t];

    // ---- load x tile (f32, fully coalesced 32B chunks), convert to f16 ----
    const float* xBase = x + (size_t)block_m * D_DIM;
#pragma unroll 4
    for (int i = 0; i < 32; ++i) {
        const int chunk = i * 256 + t;        // 8-float chunk, 128 per row
        const int row   = chunk >> 7;
        const int col   = (chunk & 127) << 3;
        v4f f0 = *(const v4f*)(xBase + (size_t)chunk * 8);
        v4f f1 = *(const v4f*)(xBase + (size_t)chunk * 8 + 4);
        v8h h;
        h[0] = (_Float16)f0[0]; h[1] = (_Float16)f0[1];
        h[2] = (_Float16)f0[2]; h[3] = (_Float16)f0[3];
        h[4] = (_Float16)f1[0]; h[5] = (_Float16)f1[1];
        h[6] = (_Float16)f1[2]; h[7] = (_Float16)f1[3];
        *(v8h*)(&ldsX[row * PITCH + col]) = h;
    }
    __syncthreads();

    // ---- x2 per row from the f16 tile (consistent rounding) ----
    {
        const int r = t >> 2, q = t & 3;
        float acc = 0.f;
        const _Float16* rp = &ldsX[r * PITCH + q * 256];
        for (int j = 0; j < 256; j += 8) {
            v8h h = *(const v8h*)(rp + j);
#pragma unroll
            for (int e = 0; e < 8; ++e) { float f = (float)h[e]; acc += f * f; }
        }
        ldsPart[t] = acc;
    }
    __syncthreads();
    if (t < M_TILE)
        ldsX2[t] = ldsPart[4*t] + ldsPart[4*t+1] + ldsPart[4*t+2] + ldsPart[4*t+3];
    __syncthreads();

    // ---- wave tiling ----
    const int wid   = t >> 5;          // 0..7
    const int lane  = t & 31;
    const int m_w   = wid >> 1;        // 4 M-subtiles of 16 rows
    const int n_w   = wid & 1;         // 2 N-subtiles of 16 cols
    const int laneN = lane & 15;
    const int kG    = lane >> 4;       // 0: lanes 0-15, 1: lanes 16-31

    // A fragment (16-bit A 16x32): lane<16 row=laneN K=kk..kk+15,
    // lane>=16 same row K=kk+16..kk+31  -> two 16B LDS loads.
    const _Float16* aRow = &ldsX[(m_w * 16 + laneN) * PITCH + kG * 16];

    float accN[8], accD[8];
#pragma unroll
    for (int v = 0; v < 8; ++v) { accN[v] = 0.f; accD[v] = 0.f; }

    const float kExp = -7.0444087e-4f;   // -log2(e) / (2 * 32^2)

    for (int nb = 0; nb < D_DIM; nb += 32) {
        const int n0  = nb + n_w * 16;
        const int col = n0 + laneN;
        // B fragment (32x16 KxN, 16-bit): lane<16 N=laneN K=kk..kk+15,
        // lane>=16 N=laneN K=kk+16..kk+31 -> one contiguous 32B load from Wt.
        const _Float16* bPtr = Wt + (size_t)col * D_DIM + kG * 16;

        v8f c = {};
#pragma unroll 4
        for (int kk = 0; kk < D_DIM; kk += 32) {
            union { v16h v; v8h h[2]; } A, B;
            A.h[0] = *(const v8h*)(aRow + kk);
            A.h[1] = *(const v8h*)(aRow + kk + 8);
            B.v    = *(const v16h*)(bPtr + kk);
            c = __builtin_amdgcn_wmma_f32_16x16x32_f16(
                    false, A.v, false, B.v, (short)0, c, false, false);
        }

        // fused epilogue: C layout VGPR v -> (M = v + 8*kG, N = laneN)
        const float p2c = ldsP2[col];
        const float sc  = ldsS[col];
#pragma unroll
        for (int v = 0; v < 8; ++v) {
            const int row = m_w * 16 + v + kG * 8;
            float d2 = ldsX2[row] + p2c - 2.0f * c[v];
            d2 = fmaxf(d2, 0.0f);
            const float b = __builtin_amdgcn_exp2f(d2 * kExp);
            accD[v] += b;
            accN[v] += b * sc;
        }
    }

    // ---- reduce over N: butterfly within each 16-lane group, then LDS ----
#pragma unroll
    for (int v = 0; v < 8; ++v) {
        float dn = accD[v], nm = accN[v];
#pragma unroll
        for (int m = 1; m < 16; m <<= 1) {
            dn += __shfl_xor(dn, m, 32);
            nm += __shfl_xor(nm, m, 32);
        }
        if (laneN == 0) {
            const int row = m_w * 16 + v + kG * 8;
            ldsDen[row][n_w] = dn;
            ldsNum[row][n_w] = nm;
        }
    }
    __syncthreads();
    if (t < M_TILE) {
        const float dn = ldsDen[t][0] + ldsDen[t][1];
        const float nm = ldsNum[t][0] + ldsNum[t][1];
        out[block_m + t] = nm / dn;
    }
}

// -------------------------------------------------------------------------
extern "C" void kernel_launch(void* const* d_in, const int* in_sizes, int n_in,
                              void* d_out, int out_size, void* d_ws, size_t ws_size,
                              hipStream_t stream)
{
    const float* x = (const float*)d_in[0];
    const float* W = (const float*)d_in[1];
    float* out = (float*)d_out;

    const int N = in_sizes[0] / D_DIM;          // 65536

    // workspace: Wt f16 (2 MB) | p2 (4 KB) | s (4 KB)
    char* ws = (char*)d_ws;
    _Float16* Wt = (_Float16*)ws;
    float* p2 = (float*)(ws + (size_t)D_DIM * D_DIM * sizeof(_Float16));
    float* s  = p2 + D_DIM;

    grnn_prep_w <<<D_DIM,     256, 0, stream>>>(W, Wt, s);
    grnn_prep_p2<<<D_DIM,     128, 0, stream>>>(Wt, p2);
    grnn_main   <<<N / M_TILE, 256, 0, stream>>>(x, Wt, p2, s, out);
}
